// CausalTransformerOneMinusOne_34823594836520
// MI455X (gfx1250) — matrix operand order repaired and checked by reading, not compiled, not run on hardware
//
#include <hip/hip_runtime.h>

typedef __attribute__((ext_vector_type(16))) __bf16 v16bf;
typedef __attribute__((ext_vector_type(8)))  __bf16 v8bf;
typedef __attribute__((ext_vector_type(8)))  float  v8f;

#define BB 16
#define SS 4096
#define DD 512
#define EE 1536   // 3*D
#define MMDIM 2048
#define ROWS (BB*SS)   // 65536

#define BM 128
#define BN 128
#define BK 32
#define NKSTEPS (DD / BK)   // 16

// ---------------------------------------------------------------------------
// Kernel 1: qkv = x @ W_qkv + b_qkv   (65536x512) x (512x1536)
// bf16x3 split for ~fp32 accuracy on v_wmma_f32_16x16x32_bf16.
// 8 waves/WG in 4(M) x 2(N); each wave: 32x64 = 2x4 fragments of 16x16.
// Double-buffered LDS (ping-pong) so tile i+1 staging overlaps tile i WMMAs.
// ---------------------------------------------------------------------------
__global__ __launch_bounds__(256) void qkv_gemm(
    const float* __restrict__ x, const float* __restrict__ Wqkv,
    const float* __restrict__ bqkv,
    float* __restrict__ qout, float* __restrict__ kout, float* __restrict__ vout)
{
    __shared__ __align__(32) __bf16 As_hi[2][BM][BK];
    __shared__ __align__(32) __bf16 As_lo[2][BM][BK];
    __shared__ __align__(32) __bf16 Bs_hi[2][BN][BK];   // stored transposed: [n][k]
    __shared__ __align__(32) __bf16 Bs_lo[2][BN][BK];

    const int tid    = threadIdx.x;
    const int lane   = tid & 31;
    const int wid    = tid >> 5;
    const int wave_m = wid >> 1;   // 0..3  (32 rows each)
    const int wave_n = wid & 1;    // 0..1  (64 cols each)
    const int nl     = lane & 15;
    const int half   = lane >> 4;

    const int mBase = blockIdx.y * BM;
    const int nBase = blockIdx.x * BN;

    const v8f vzero = {0.f,0.f,0.f,0.f,0.f,0.f,0.f,0.f};
    v8f acc[2][4];
    #pragma unroll
    for (int i = 0; i < 2; ++i)
        #pragma unroll
        for (int j = 0; j < 4; ++j) acc[i][j] = vzero;

    // loader thread mapping
    const int a_c4 = tid & 7;    // 8 float4 per 32-col A row  -> 32 rows/pass x4
    const int a_r  = tid >> 3;
    const int b_c4 = tid & 31;   // 32 float4 per 128-col B row -> 8 rows/pass x4
    const int b_r  = tid >> 5;

    float4 rA[4], rB[4];

    auto load_global = [&](int k0) {
        #pragma unroll
        for (int p = 0; p < 4; ++p)
            rA[p] = *(const float4*)(x + (size_t)(mBase + p * 32 + a_r) * DD + k0 + a_c4 * 4);
        #pragma unroll
        for (int p = 0; p < 4; ++p)
            rB[p] = *(const float4*)(Wqkv + (size_t)(k0 + p * 8 + b_r) * EE + nBase + b_c4 * 4);
    };

    auto stage_lds = [&](int buf) {
        #pragma unroll
        for (int p = 0; p < 4; ++p) {
            const float f[4] = {rA[p].x, rA[p].y, rA[p].z, rA[p].w};
            #pragma unroll
            for (int j = 0; j < 4; ++j) {
                const __bf16 h = (__bf16)f[j];
                const __bf16 l = (__bf16)(f[j] - (float)h);
                As_hi[buf][p * 32 + a_r][a_c4 * 4 + j] = h;
                As_lo[buf][p * 32 + a_r][a_c4 * 4 + j] = l;
            }
        }
        #pragma unroll
        for (int p = 0; p < 4; ++p) {
            const float f[4] = {rB[p].x, rB[p].y, rB[p].z, rB[p].w};
            #pragma unroll
            for (int j = 0; j < 4; ++j) {
                const __bf16 h = (__bf16)f[j];
                const __bf16 l = (__bf16)(f[j] - (float)h);
                Bs_hi[buf][b_c4 * 4 + j][p * 8 + b_r] = h;
                Bs_lo[buf][b_c4 * 4 + j][p * 8 + b_r] = l;
            }
        }
    };

    // prologue: stage tile 0
    load_global(0);
    stage_lds(0);
    __syncthreads();

    for (int i = 0; i < NKSTEPS; ++i) {
        const int cur = i & 1;
        if (i + 1 < NKSTEPS) load_global((i + 1) * BK);   // VMEM in flight under WMMAs

        // A fragments resident for both M-frags (contiguous k order used here:
        // element e -> K = 16*half + e, row = nl)
        v16bf aHi[2], aLo[2];
        #pragma unroll
        for (int mf = 0; mf < 2; ++mf) {
            const int arow = wave_m * 32 + mf * 16 + nl;
            aHi[mf] = *(const v16bf*)&As_hi[cur][arow][half * 16];
            aLo[mf] = *(const v16bf*)&As_lo[cur][arow][half * 16];
        }
        #pragma unroll
        for (int nf = 0; nf < 4; ++nf) {
            // B fragment (contiguous k order, same as A): element e -> K = 16*half + e, col = nl
            const int bcol = wave_n * 64 + nf * 16 + nl;
            const v16bf bHi = *(const v16bf*)&Bs_hi[cur][bcol][half * 16];
            const v16bf bLo = *(const v16bf*)&Bs_lo[cur][bcol][half * 16];
            #pragma unroll
            for (int mf = 0; mf < 2; ++mf) {
                acc[mf][nf] = __builtin_amdgcn_wmma_f32_16x16x32_bf16(
                    false, aHi[mf], false, bHi, (short)0, acc[mf][nf], false, false);
                acc[mf][nf] = __builtin_amdgcn_wmma_f32_16x16x32_bf16(
                    false, aHi[mf], false, bLo, (short)0, acc[mf][nf], false, false);
                acc[mf][nf] = __builtin_amdgcn_wmma_f32_16x16x32_bf16(
                    false, aLo[mf], false, bHi, (short)0, acc[mf][nf], false, false);
            }
        }

        if (i + 1 < NKSTEPS) stage_lds(cur ^ 1);   // write the OTHER buffer
        __syncthreads();
    }

    // --- epilogue: bias + scatter to q / k / v (each 16-col frag is in one segment) ---
    #pragma unroll
    for (int mf = 0; mf < 2; ++mf) {
        #pragma unroll
        for (int nf = 0; nf < 4; ++nf) {
            const int gcol = nBase + wave_n * 64 + nf * 16 + nl;
            const int seg  = gcol >> 9;        // 0:q 1:k 2:v
            const int cin  = gcol & 511;
            const float bias = bqkv[gcol];
            float* dst = (seg == 0) ? qout : (seg == 1) ? kout : vout;
            const int gm0 = mBase + wave_m * 32 + mf * 16 + half * 8;  // C: M = r + 8*half
            #pragma unroll
            for (int r = 0; r < 8; ++r)
                dst[(size_t)(gm0 + r) * DD + cin] = acc[mf][nf][r] + bias;
        }
    }
}

// ---------------------------------------------------------------------------
// Kernel 2: scores[b][s] = dot(q[b][S-1], k[b][s])   (one wave per row)
// ---------------------------------------------------------------------------
__global__ __launch_bounds__(256) void scores_kernel(
    const float* __restrict__ q, const float* __restrict__ k, float* __restrict__ sc)
{
    const int lane = threadIdx.x & 31;
    const int wid  = threadIdx.x >> 5;
    const int row  = blockIdx.x * 8 + wid;       // 0..65535
    const int b    = row >> 12;
    const float* qrow = q + (size_t)(b * SS + (SS - 1)) * DD;
    const float* krow = k + (size_t)row * DD;
    float p = 0.f;
    #pragma unroll
    for (int d = lane * 4; d < DD; d += 32 * 4) {
        const float4 kv = *(const float4*)(krow + d);
        const float4 qv = *(const float4*)(qrow + d);
        p += kv.x * qv.x + kv.y * qv.y + kv.z * qv.z + kv.w * qv.w;
    }
    #pragma unroll
    for (int off = 16; off > 0; off >>= 1) p += __shfl_xor(p, off, 32);
    if (lane == 0) sc[row] = p;
}

// ---------------------------------------------------------------------------
// Kernel 3: in-place softmax over s for each b
// ---------------------------------------------------------------------------
__global__ __launch_bounds__(256) void softmax_kernel(float* __restrict__ sc)
{
    __shared__ float red[256];
    const int b = blockIdx.x;
    const int tid = threadIdx.x;
    float* row = sc + (size_t)b * SS;

    float m = -3.4e38f;
    for (int i = tid; i < SS; i += 256) m = fmaxf(m, row[i]);
    red[tid] = m; __syncthreads();
    for (int s = 128; s > 0; s >>= 1) {
        if (tid < s) red[tid] = fmaxf(red[tid], red[tid + s]);
        __syncthreads();
    }
    const float mx = red[0];
    __syncthreads();

    float ssum = 0.f;
    for (int i = tid; i < SS; i += 256) {
        const float e = __expf(row[i] - mx);
        row[i] = e;
        ssum += e;
    }
    red[tid] = ssum; __syncthreads();
    for (int s = 128; s > 0; s >>= 1) {
        if (tid < s) red[tid] += red[tid + s];
        __syncthreads();
    }
    const float inv = 1.f / red[0];
    for (int i = tid; i < SS; i += 256) row[i] *= inv;
}

// ---------------------------------------------------------------------------
// Kernel 4: partial[b][chunk][d] = sum_{s in chunk} attn[b][s] * v[b][s][d]
// (deterministic two-stage reduction; no float atomics)
// ---------------------------------------------------------------------------
__global__ __launch_bounds__(256) void attnv_partial(
    const float* __restrict__ attn, const float* __restrict__ v, float* __restrict__ part)
{
    const int b  = blockIdx.x >> 4;
    const int ch = blockIdx.x & 15;
    const int tid = threadIdx.x;
    float a0 = 0.f, a1 = 0.f;
    for (int s = ch * 256; s < ch * 256 + 256; ++s) {
        const float w = attn[b * SS + s];
        const float* vr = v + (size_t)(b * SS + s) * DD;
        a0 += w * vr[tid];
        a1 += w * vr[tid + 256];
    }
    part[(size_t)blockIdx.x * DD + tid]       = a0;
    part[(size_t)blockIdx.x * DD + tid + 256] = a1;
}

// ---------------------------------------------------------------------------
// Kernel 5: h[b][d] = x[b][S-1][d] + sum_c partial[b][c][d]
// ---------------------------------------------------------------------------
__global__ __launch_bounds__(256) void h_residual(
    const float* __restrict__ x, const float* __restrict__ part, float* __restrict__ h)
{
    const int idx = blockIdx.x * 256 + threadIdx.x;  // 0..8191
    const int b = idx >> 9;
    const int d = idx & 511;
    float acc = x[(size_t)(b * SS + SS - 1) * DD + d];
    #pragma unroll
    for (int c = 0; c < 16; ++c) acc += part[(size_t)(b * 16 + c) * DD + d];
    h[idx] = acc;
}

// ---------------------------------------------------------------------------
// Kernels 6-8: tiny MLP (fp32 VALU; memory-bound on weights via L2)
// ---------------------------------------------------------------------------
__global__ __launch_bounds__(256) void mlp1(
    const float* __restrict__ h, const float* __restrict__ W1,
    const float* __restrict__ b1, float* __restrict__ h1)
{
    const int idx = blockIdx.x * 256 + threadIdx.x;   // 0..32767
    const int b = idx >> 11;
    const int col = idx & 2047;
    float acc = b1[col];
    const float* hr = h + b * DD;
    for (int d = 0; d < DD; ++d) acc += hr[d] * W1[(size_t)d * MMDIM + col];
    h1[idx] = fmaxf(acc, 0.f);
}

__global__ __launch_bounds__(256) void mlp2(
    const float* __restrict__ h1, const float* __restrict__ W2,
    const float* __restrict__ b2, float* __restrict__ h2)
{
    const int idx = blockIdx.x * 256 + threadIdx.x;   // 0..32767
    const int b = idx >> 11;
    const int col = idx & 2047;
    float acc = b2[col];
    const float* hr = h1 + b * MMDIM;
    for (int d = 0; d < MMDIM; ++d) acc += hr[d] * W2[(size_t)d * MMDIM + col];
    h2[idx] = fmaxf(acc, 0.f);
}

__global__ __launch_bounds__(512) void mlp3(
    const float* __restrict__ h2, const float* __restrict__ W3,
    const float* __restrict__ b3, float* __restrict__ out)
{
    const int lane = threadIdx.x & 31;
    const int b = threadIdx.x >> 5;   // 16 waves, one per batch row
    float p = 0.f;
    for (int d = lane; d < MMDIM; d += 32) p += h2[b * MMDIM + d] * W3[d];
    #pragma unroll
    for (int off = 16; off > 0; off >>= 1) p += __shfl_xor(p, off, 32);
    if (lane == 0) out[b] = p + b3[0];
}

// ---------------------------------------------------------------------------
extern "C" void kernel_launch(void* const* d_in, const int* in_sizes, int n_in,
                              void* d_out, int out_size, void* d_ws, size_t ws_size,
                              hipStream_t stream)
{
    (void)in_sizes; (void)n_in; (void)out_size; (void)ws_size;

    const float* x    = (const float*)d_in[0];
    const float* Wqkv = (const float*)d_in[1];
    const float* bqkv = (const float*)d_in[2];
    const float* W1   = (const float*)d_in[3];
    const float* b1   = (const float*)d_in[4];
    const float* W2   = (const float*)d_in[5];
    const float* b2   = (const float*)d_in[6];
    const float* W3   = (const float*)d_in[7];
    const float* b3   = (const float*)d_in[8];

    // d_out = [out(16) | q(B*S*D) | k(B*S*D)]
    float* out  = (float*)d_out;
    float* qout = out + 16;
    float* kout = qout + (size_t)ROWS * DD;

    // workspace: v | scores/attn | partial | h | h1 | h2
    float* ws   = (float*)d_ws;
    float* vws  = ws;                               // ROWS*DD
    float* sc   = vws + (size_t)ROWS * DD;          // ROWS
    float* part = sc + ROWS;                        // BB*16*DD
    float* h    = part + (size_t)BB * 16 * DD;      // BB*DD
    float* h1   = h + (size_t)BB * DD;              // BB*MMDIM
    float* h2   = h1 + (size_t)BB * MMDIM;          // BB*MMDIM

    qkv_gemm<<<dim3(EE / BN, ROWS / BM), 256, 0, stream>>>(x, Wqkv, bqkv, qout, kout, vws);
    scores_kernel<<<ROWS / 8, 256, 0, stream>>>(qout, kout, sc);
    softmax_kernel<<<BB, 256, 0, stream>>>(sc);
    attnv_partial<<<BB * 16, 256, 0, stream>>>(sc, vws, part);
    h_residual<<<BB * DD / 256, 256, 0, stream>>>(x, part, h);
    mlp1<<<BB * MMDIM / 256, 256, 0, stream>>>(h, W1, b1, h1);
    mlp2<<<BB * MMDIM / 256, 256, 0, stream>>>(h1, W2, b2, h2);
    mlp3<<<1, 512, 0, stream>>>(h2, W3, b3, out);
}
